// QuerySchemaEncoder_5471788335920
// MI455X (gfx1250) — matrix operand order repaired and checked by reading, not compiled, not run on hardware
//
#include <hip/hip_runtime.h>
#include <hip/hip_bf16.h>

// Problem constants (match reference)
#define BB 4096
#define CC 64
#define LQ 32
#define DD 128
#define HH 192
#define K1 256   // 2*D
#define DG 512   // 4*D

typedef __attribute__((ext_vector_type(16))) __bf16    v16bf;
typedef __attribute__((ext_vector_type(8)))  float     v8f;
typedef __attribute__((ext_vector_type(8)))  unsigned  v8u;

// ---- bf16 helpers ----
// RNE conversion: used only where values are *stored* for later reuse.
__device__ __forceinline__ __bf16 f2bf(float f) {
    unsigned u = __builtin_bit_cast(unsigned, f);
    unsigned r = (u + 0x7FFFu + ((u >> 16) & 1u)) >> 16;
    unsigned short s = (unsigned short)r;
    return __builtin_bit_cast(__bf16, s);
}
__device__ __forceinline__ float bf2f(__bf16 b) {
    unsigned short s = __builtin_bit_cast(unsigned short, b);
    unsigned u = ((unsigned)s) << 16;
    return __builtin_bit_cast(float, u);
}
// Truncating pack of two f32 into one dword of two bf16 (lowers to v_perm).
__device__ __forceinline__ unsigned pack2bf(float f0, float f1) {
    unsigned u0 = __builtin_bit_cast(unsigned, f0);
    unsigned u1 = __builtin_bit_cast(unsigned, f1);
    return (u0 >> 16) | (u1 & 0xFFFF0000u);
}

// =====================================================================
// Weight packing: W (f32, [K,N] row-major, two sources split at `split`)
// -> bf16 B-fragment layout: F[((ntile*ksteps + kstep)*32 + lane)*16 + e]
// with n = ntile*16 + (lane&15), k = kstep*32 + (lane>>4)*8 + (e&7) + (e>>3)*16
// =====================================================================
__global__ void pack_b2_kernel(const float* __restrict__ Wa,
                               const float* __restrict__ Wb,
                               int split, int K, int N,
                               __bf16* __restrict__ F) {
    int total = K * N;
    int ksteps = K >> 5;
    for (int idx = blockIdx.x * blockDim.x + threadIdx.x; idx < total;
         idx += gridDim.x * blockDim.x) {
        int e    = idx & 15;
        int lane = (idx >> 4) & 31;
        int rest = idx >> 9;             // ntile*ksteps + kstep
        int kstep = rest % ksteps;
        int ntile = rest / ksteps;
        int n = (ntile << 4) + (lane & 15);
        int k = (kstep << 5) + ((lane >> 4) << 3) + (e & 7) + ((e >> 3) << 4);
        float w = (k < split) ? Wa[(size_t)k * N + n]
                              : Wb[(size_t)(k - split) * N + n];
        F[idx] = f2bf(w);
    }
}

// Zero-init LSTM state (h, c f32; h bf16 copy)
__global__ void init_state_kernel(float* __restrict__ h, float* __restrict__ c,
                                  __bf16* __restrict__ hb) {
    int idx = blockIdx.x * blockDim.x + threadIdx.x;
    if (idx < BB * DD) {
        h[idx] = 0.0f;
        c[idx] = 0.0f;
        hb[idx] = f2bf(0.0f);
    }
}

// ---- fragment loaders ----
// A fragment, f32 source (p = row base + k offset for this lane):
// lane&15 = M, kbase=(lane>>4)*8; elems 0..7 -> K=kbase+e, 8..15 -> K=kbase+16+(e-8)
__device__ __forceinline__ v16bf load_a_f32(const float* __restrict__ p) {
    v8u u;
#pragma unroll
    for (int i = 0; i < 4; i++)
        u[i] = pack2bf(p[2 * i], p[2 * i + 1]);
#pragma unroll
    for (int i = 0; i < 4; i++)
        u[4 + i] = pack2bf(p[16 + 2 * i], p[17 + 2 * i]);
    return __builtin_bit_cast(v16bf, u);
}
__device__ __forceinline__ v16bf load_a_bf16(const __bf16* __restrict__ p) {
    v16bf a;
#pragma unroll
    for (int e = 0; e < 8; e++) a[e] = p[e];
#pragma unroll
    for (int e = 0; e < 8; e++) a[8 + e] = p[16 + e];
    return a;
}

__device__ __forceinline__ v8f wmma_bf16(v16bf a, v16bf b, v8f c) {
    return __builtin_amdgcn_wmma_f32_16x16x32_bf16(
        /*neg_a=*/false, a, /*neg_b=*/false, b,
        /*c_mod=*/(short)0, c, /*reuse_a=*/false, /*reuse_b=*/false);
}

// =====================================================================
// GEMM1: X[262144 x 256] (concat header|table_word) @ W1 -> relu -> Hbuf bf16
// One wave per M-tile; all 12 N-tiles register-blocked (A read once).
// B fragments batched into a live register array per K-step so loads
// issue as one clause and wmmas overlap remaining loads.
// =====================================================================
__global__ void table_gemm1_kernel(const float* __restrict__ hdr,
                                   const float* __restrict__ tw,
                                   const __bf16* __restrict__ W1f,
                                   const float* __restrict__ b1,
                                   __bf16* __restrict__ Hbuf) {
    const int NT = HH / 16;                       // 12
    int lane  = threadIdx.x & 31;
    int wave  = threadIdx.x >> 5;
    int mtile = blockIdx.x * 8 + wave;            // 16384 m-tiles
    int m  = (mtile << 4) + (lane & 15);
    int kb = ((lane >> 4) << 3);

    v8f acc[NT];
#pragma unroll
    for (int nt = 0; nt < NT; nt++) {
        float bias = b1[(nt << 4) + (lane & 15)];
#pragma unroll
        for (int r = 0; r < 8; r++) acc[nt][r] = bias;
    }

#pragma unroll
    for (int ks = 0; ks < 8; ks++) {
        const float* src = (ks < 4) ? hdr : tw;
        int kk = ((ks & 3) << 5) + kb;
        v16bf a = load_a_f32(src + (size_t)m * DD + kk);
        v16bf bf[NT];
#pragma unroll
        for (int nt = 0; nt < NT; nt++)
            bf[nt] = *(const v16bf*)(W1f + (((size_t)(nt * 8 + ks) * 32 + lane) << 4));
#pragma unroll
        for (int nt = 0; nt < NT; nt++)
            acc[nt] = wmma_bf16(a, bf[nt], acc[nt]);
    }

    int mrow0 = (mtile << 4) + ((lane >> 4) << 3);
#pragma unroll
    for (int nt = 0; nt < NT; nt++) {
        int n = (nt << 4) + (lane & 15);
#pragma unroll
        for (int r = 0; r < 8; r++) {
            float v = acc[nt][r];
            v = v > 0.0f ? v : 0.0f;
            Hbuf[(size_t)(mrow0 + r) * HH + n] = f2bf(v);
        }
    }
}

// =====================================================================
// GEMM2: Hbuf[262144 x 192] @ W2 -> relu -> TE bf16 [M x 128]
// One wave per M-tile; all 8 N-tiles register-blocked; batched B loads.
// =====================================================================
__global__ void table_gemm2_kernel(const __bf16* __restrict__ Hbuf,
                                   const __bf16* __restrict__ W2f,
                                   const float* __restrict__ b2,
                                   __bf16* __restrict__ TE) {
    const int NT = DD / 16;                       // 8
    const int KS = HH / 32;                       // 6
    int lane  = threadIdx.x & 31;
    int wave  = threadIdx.x >> 5;
    int mtile = blockIdx.x * 8 + wave;
    int m  = (mtile << 4) + (lane & 15);
    int kb = ((lane >> 4) << 3);

    v8f acc[NT];
#pragma unroll
    for (int nt = 0; nt < NT; nt++) {
        float bias = b2[(nt << 4) + (lane & 15)];
#pragma unroll
        for (int r = 0; r < 8; r++) acc[nt][r] = bias;
    }

#pragma unroll
    for (int ks = 0; ks < KS; ks++) {
        int kk = (ks << 5) + kb;
        v16bf a = load_a_bf16(Hbuf + (size_t)m * HH + kk);
        v16bf bf[NT];
#pragma unroll
        for (int nt = 0; nt < NT; nt++)
            bf[nt] = *(const v16bf*)(W2f + (((size_t)(nt * KS + ks) * 32 + lane) << 4));
#pragma unroll
        for (int nt = 0; nt < NT; nt++)
            acc[nt] = wmma_bf16(a, bf[nt], acc[nt]);
    }

    int mrow0 = (mtile << 4) + ((lane >> 4) << 3);
#pragma unroll
    for (int nt = 0; nt < NT; nt++) {
        int n = (nt << 4) + (lane & 15);
#pragma unroll
        for (int r = 0; r < 8; r++) {
            float v = acc[nt][r];
            v = v > 0.0f ? v : 0.0f;
            TE[(size_t)(mrow0 + r) * DD + n] = f2bf(v);
        }
    }
}

// =====================================================================
// Masked mean over C + l2norm -> te_final f32 [B x 128]; one block per b
// =====================================================================
__global__ void te_reduce_kernel(const __bf16* __restrict__ TE,
                                 const int* __restrict__ ncols,
                                 float* __restrict__ te_final) {
    int b = blockIdx.x;
    int d = threadIdx.x;                          // 128 threads
    int nc = ncols[b];
    const __bf16* p = TE + (size_t)b * CC * DD + d;
    float s = 0.0f;
    for (int c = 0; c < nc; c++) s += bf2f(p[(size_t)c * DD]);
    float v = s / (float)nc;

    __shared__ float sm[DD];
    sm[d] = v * v;
    __syncthreads();
#pragma unroll
    for (int k = 64; k > 0; k >>= 1) {
        if (d < k) sm[d] += sm[d + k];
        __syncthreads();
    }
    float rs = rsqrtf(fmaxf(sm[0], 1e-12f));
    te_final[(size_t)b * DD + d] = v * rs;
}

// =====================================================================
// LSTM step GEMM: z = [x_t | h] @ [Wx;Wh] + b_lstm -> z f32 [B x 512]
// Wave handles one M-tile and half of N (16 N-tiles), processed as two
// chunks of 8 per K-step to bound register pressure while keeping
// 8-deep batched B loads.
// =====================================================================
__global__ void lstm_gemm_kernel(const float* __restrict__ q,
                                 const __bf16* __restrict__ hb,
                                 const __bf16* __restrict__ Wcf,
                                 const float* __restrict__ bl,
                                 float* __restrict__ z, int t) {
    const int NTH = 16;                           // n-tiles per wave (of 32)
    const int CH  = 8;                            // chunk size
    int lane = threadIdx.x & 31;
    int wave = threadIdx.x >> 5;
    int tid  = blockIdx.x * 8 + wave;             // 512 waves
    int mtile = tid >> 1;
    int nb    = (tid & 1) * NTH;                  // n-tile base
    int m  = (mtile << 4) + (lane & 15);
    int kb = ((lane >> 4) << 3);

    v8f acc[NTH];
#pragma unroll
    for (int nt = 0; nt < NTH; nt++) {
        float bias = bl[((nb + nt) << 4) + (lane & 15)];
#pragma unroll
        for (int r = 0; r < 8; r++) acc[nt][r] = bias;
    }

#pragma unroll
    for (int ks = 0; ks < 8; ks++) {
        v16bf a;
        if (ks < 4) {
            int kk = (ks << 5) + kb;
            a = load_a_f32(q + ((size_t)m * LQ + t) * DD + kk);
        } else {
            int kk = ((ks - 4) << 5) + kb;
            a = load_a_bf16(hb + (size_t)m * DD + kk);
        }
#pragma unroll
        for (int ch = 0; ch < NTH / CH; ch++) {
            v16bf bf[CH];
#pragma unroll
            for (int j = 0; j < CH; j++) {
                int nt = ch * CH + j;
                bf[j] = *(const v16bf*)(Wcf + (((size_t)((nb + nt) * 8 + ks) * 32 + lane) << 4));
            }
#pragma unroll
            for (int j = 0; j < CH; j++) {
                int nt = ch * CH + j;
                acc[nt] = wmma_bf16(a, bf[j], acc[nt]);
            }
        }
    }

    int mrow0 = (mtile << 4) + ((lane >> 4) << 3);
#pragma unroll
    for (int nt = 0; nt < NTH; nt++) {
        int n = ((nb + nt) << 4) + (lane & 15);
#pragma unroll
        for (int r = 0; r < 8; r++) {
            z[(size_t)(mrow0 + r) * DG + n] = acc[nt][r];
        }
    }
}

// =====================================================================
// LSTM elementwise update with length masking
// =====================================================================
__global__ void lstm_update_kernel(const float* __restrict__ z,
                                   const int* __restrict__ qlen,
                                   float* __restrict__ h, float* __restrict__ c,
                                   __bf16* __restrict__ hb, int t) {
    int idx = blockIdx.x * blockDim.x + threadIdx.x;
    if (idx >= BB * DD) return;
    int b = idx >> 7;
    int d = idx & (DD - 1);
    const float* zr = z + (size_t)b * DG;
    float zi = zr[d];
    float zf = zr[DD + d];
    float zg = zr[2 * DD + d];
    float zo = zr[3 * DD + d];
    float gi = 1.0f / (1.0f + __expf(-zi));
    float gf = 1.0f / (1.0f + __expf(-zf));
    float gg = zg > 0.0f ? zg : 0.0f;             // activation='relu'
    float go = 1.0f / (1.0f + __expf(-zo));
    float cn = gf * c[idx] + gi * gg;
    float rc = cn > 0.0f ? cn : 0.0f;
    float hn = go * rc;
    bool msk = t < qlen[b];
    float hv = msk ? hn : h[idx];
    float cv = msk ? cn : c[idx];
    h[idx] = hv;
    c[idx] = cv;
    hb[idx] = f2bf(hv);
}

// =====================================================================
// Per-row loss: l2norm(qe), squared distance to te, contrastive terms
// =====================================================================
__global__ void loss_kernel(const float* __restrict__ h,
                            const float* __restrict__ te_final,
                            const float* __restrict__ labels,
                            float* __restrict__ lossbuf) {
    int b = blockIdx.x;
    int d = threadIdx.x;                          // 128 threads
    float hv = h[(size_t)b * DD + d];

    __shared__ float sm[DD];
    sm[d] = hv * hv;
    __syncthreads();
#pragma unroll
    for (int k = 64; k > 0; k >>= 1) {
        if (d < k) sm[d] += sm[d + k];
        __syncthreads();
    }
    float rs = rsqrtf(fmaxf(sm[0], 1e-12f));
    __syncthreads();

    float qv = hv * rs;
    float diff = qv - te_final[(size_t)b * DD + d];
    sm[d] = diff * diff;
    __syncthreads();
#pragma unroll
    for (int k = 64; k > 0; k >>= 1) {
        if (d < k) sm[d] += sm[d + k];
        __syncthreads();
    }
    if (d == 0) {
        float dist = sm[0];
        float lab = labels[b];
        float mterm = 1.0f - sqrtf(dist);         // MARGIN = 1.0
        mterm = mterm > 0.0f ? mterm : 0.0f;
        lossbuf[b] = lab * dist + (1.0f - lab) * mterm * mterm;
    }
}

__global__ void final_reduce_kernel(const float* __restrict__ lossbuf,
                                    float* __restrict__ out) {
    __shared__ float sm[256];
    float s = 0.0f;
    for (int i = threadIdx.x; i < BB; i += 256) s += lossbuf[i];
    sm[threadIdx.x] = s;
    __syncthreads();
#pragma unroll
    for (int k = 128; k > 0; k >>= 1) {
        if (threadIdx.x < k) sm[threadIdx.x] += sm[threadIdx.x + k];
        __syncthreads();
    }
    if (threadIdx.x == 0) out[0] = 0.5f * sm[0] / (float)BB;
}

// =====================================================================
// Host launcher
// =====================================================================
extern "C" void kernel_launch(void* const* d_in, const int* in_sizes, int n_in,
                              void* d_out, int out_size, void* d_ws, size_t ws_size,
                              hipStream_t stream) {
    (void)in_sizes; (void)n_in; (void)out_size; (void)ws_size;

    const float* q      = (const float*)d_in[0];   // [B, LQ, D]
    const int*   qlen   = (const int*)  d_in[1];   // [B]
    const float* hdr    = (const float*)d_in[2];   // [B, C, D]
    const float* tw     = (const float*)d_in[3];   // [B, C, D]
    const int*   ncols  = (const int*)  d_in[4];   // [B]
    // d_in[5] = masks (recomputed from num_cols)
    const float* labels = (const float*)d_in[6];   // [B]
    const float* W1     = (const float*)d_in[7];   // [256,192]
    const float* b1     = (const float*)d_in[8];   // [192]
    const float* W2     = (const float*)d_in[9];   // [192,128]
    const float* b2     = (const float*)d_in[10];  // [128]
    const float* Wx     = (const float*)d_in[11];  // [128,512]
    const float* Wh     = (const float*)d_in[12];  // [128,512]
    const float* bl     = (const float*)d_in[13];  // [512]
    float* out = (float*)d_out;

    // Workspace layout (bytes)
    char* ws = (char*)d_ws;
    size_t off = 0;
    __bf16* W1f  = (__bf16*)(ws + off); off += (size_t)K1 * HH * 2;        // 98304
    __bf16* W2f  = (__bf16*)(ws + off); off += (size_t)HH * DD * 2;        // 49152
    __bf16* Wcf  = (__bf16*)(ws + off); off += (size_t)K1 * DG * 2;        // 262144
    __bf16* Hbuf = (__bf16*)(ws + off); off += (size_t)BB * CC * HH * 2;   // ~100.7 MB
    __bf16* TE   = (__bf16*)(ws + off); off += (size_t)BB * CC * DD * 2;   // ~67.1 MB
    float*  teF  = (float*) (ws + off); off += (size_t)BB * DD * 4;        // 2 MB
    float*  hS   = (float*) (ws + off); off += (size_t)BB * DD * 4;        // 2 MB
    float*  cS   = (float*) (ws + off); off += (size_t)BB * DD * 4;        // 2 MB
    __bf16* hB   = (__bf16*)(ws + off); off += (size_t)BB * DD * 2;        // 1 MB
    float*  zS   = (float*) (ws + off); off += (size_t)BB * DG * 4;        // 8 MB
    float*  lb   = (float*) (ws + off); off += (size_t)BB * 4;             // 16 KB

    // --- weight packing + state init ---
    pack_b2_kernel<<<256, 256, 0, stream>>>(W1, W1, K1, K1, HH, W1f);
    pack_b2_kernel<<<256, 256, 0, stream>>>(W2, W2, HH, HH, DD, W2f);
    pack_b2_kernel<<<256, 256, 0, stream>>>(Wx, Wh, DD, K1, DG, Wcf);
    init_state_kernel<<<(BB * DD + 255) / 256, 256, 0, stream>>>(hS, cS, hB);

    // --- table path ---
    {
        int mtiles = BB * CC / 16;                 // 16384 m-tiles
        table_gemm1_kernel<<<mtiles / 8, 256, 0, stream>>>(hdr, tw, W1f, b1, Hbuf);
        table_gemm2_kernel<<<mtiles / 8, 256, 0, stream>>>(Hbuf, W2f, b2, TE);
        te_reduce_kernel<<<BB, DD, 0, stream>>>(TE, ncols, teF);
    }

    // --- query path: 32 sequential LSTM steps ---
    {
        int wavesL = (BB / 16) * 2;                // 512 waves (2 N-halves)
        for (int t = 0; t < LQ; t++) {
            lstm_gemm_kernel<<<wavesL / 8, 256, 0, stream>>>(q, hB, Wcf, bl, zS, t);
            lstm_update_kernel<<<(BB * DD + 255) / 256, 256, 0, stream>>>(zS, qlen, hS, cS, hB, t);
        }
    }

    // --- loss ---
    loss_kernel<<<BB, DD, 0, stream>>>(hS, teF, labels, lb);
    final_reduce_kernel<<<1, 256, 0, stream>>>(lb, out);
}
